// GRUBaseline_18846316495575
// MI455X (gfx1250) — compile-verified, matrix operand order checked by reading, not run on hardware
//
#include <hip/hip_runtime.h>
#include <hip/hip_bf16.h>

#define B_   256
#define T_   512
#define E_   128
#define H_   256

typedef __attribute__((ext_vector_type(16))) __bf16 v16bf;
typedef __attribute__((ext_vector_type(8)))  float  v8f;

union AF { v16bf v; uint4 q[2]; };

__device__ __forceinline__ unsigned short f2bf_bits(float f) {
  union { float f; unsigned u; } v; v.f = f;
  unsigned r = v.u + 0x7fffu + ((v.u >> 16) & 1u);   // RNE
  return (unsigned short)(r >> 16);
}
__device__ __forceinline__ __bf16 f2bf(float f) {
  union { unsigned short s; __bf16 b; } u; u.s = f2bf_bits(f); return u.b;
}
__device__ __forceinline__ float bf2f(unsigned short s) {
  union { unsigned u; float f; } v; v.u = ((unsigned)s) << 16; return v.f;
}
__device__ __forceinline__ float sigm(float x) { return 1.0f / (1.0f + __expf(-x)); }

// Dynamic-LDS layout (272 KB total):
//   [0      , 8192  )  s_h   : bf16 hidden state      [16][256]
//   [8192   , 16384 )  s_x   : bf16 x tile, dbl buf   [2][16][128]
//   [16384  , 147456)  s_bn  : n-gate h-part B-frags  [tile16][kf8][lane32][32B]
//   [147456 , 278528)  s_brzx: r,z x-part B-frags     [tile16][g2][kk4][lane32][32B]
#define LDS_BYTES 278528

// Persistent bidirectional GRU scan.
// grid = 32 blocks: blockIdx [0,16) fwd, [16,32) bwd; 16 batch rows each.
// 8 waves/block; wave w owns hidden-column tiles w and w+8 (cols 16w, 16w+128)
// for gates r,z,n.  Per-wave registers (~450 <= 512 budget at 2 waves/SIMD):
//   r,z hidden-part weights (256 VGPRs) + n x-part (64) + 8 accums (64)
__global__ __launch_bounds__(256, 1) void gru_scan_kernel(
    const int* __restrict__ seqs, const int* __restrict__ lens,
    const float* __restrict__ embed,
    const float* __restrict__ Wih_f, const float* __restrict__ Whh_f,
    const float* __restrict__ bih_f, const float* __restrict__ bhh_f,
    const float* __restrict__ Wih_b, const float* __restrict__ Whh_b,
    const float* __restrict__ bih_b, const float* __restrict__ bhh_b,
    float* __restrict__ hcat)
{
  extern __shared__ char smem[];
  unsigned short* s_h   = (unsigned short*)smem;             // [16][256]
  unsigned short* s_x   = (unsigned short*)(smem + 8192);    // [2][16][128]
  uint4*          s_bn  = (uint4*)(smem + 16384);            // [16][8][32][2]
  uint4*          s_brzx= (uint4*)(smem + 147456);           // [16][2][4][32][2]

  const int tid  = threadIdx.x;
  const int wave = tid >> 5, lane = tid & 31;
  const int half = lane >> 4, nc = lane & 15;     // nc = N(B/C/D) = M(A)
  const int wg   = blockIdx.x & 15, dir = blockIdx.x >> 4;
  const int r0   = wg * 16;

  const float* Wih = dir ? Wih_b : Wih_f;         // [768][128]
  const float* Whh = dir ? Whh_b : Whh_f;         // [768][256]
  const float* bih = dir ? bih_b : bih_f;
  const float* bhh = dir ? bhh_b : bhh_f;

  // per-tile biases (tile t2: hidden col j = 16*(wave+8*t2)+nc)
  float biasR[2], biasZ[2], bI_n[2], bH_n[2];
#pragma unroll
  for (int t2 = 0; t2 < 2; ++t2) {
    const int j = 16 * (wave + 8 * t2) + nc;
    biasR[t2] = bih[j]        + bhh[j];
    biasZ[t2] = bih[H_ + j]   + bhh[H_ + j];
    bI_n[t2]  = bih[2*H_ + j];
    bH_n[t2]  = bhh[2*H_ + j];
  }

  // ---- build weight fragments ----
  // element i of v16bf <-> K = 32*kf + 8*half + i + (i<8?0:8)
  v16bf Brh[2][8], Bzh[2][8], Bnx[2][4];
#pragma unroll
  for (int t2 = 0; t2 < 2; ++t2) {
    const int ct = wave + 8 * t2;                 // column tile 0..15
    const int j  = 16 * ct + nc;
    const int rowR = j, rowZ = H_ + j, rowN = 2*H_ + j;
#pragma unroll
    for (int kf = 0; kf < 12; ++kf) {
      v16bf fr = {}, fz = {}, fn = {};
#pragma unroll
      for (int i = 0; i < 16; ++i) {
        const int k = 32*kf + 8*half + i + ((i < 8) ? 0 : 8);
        float wr, wz, wn;
        if (k < H_) {
          wr = Whh[rowR * H_ + k]; wz = Whh[rowZ * H_ + k]; wn = Whh[rowN * H_ + k];
        } else {
          const int kx = k - H_;
          wr = Wih[rowR * E_ + kx]; wz = Wih[rowZ * E_ + kx]; wn = Wih[rowN * E_ + kx];
        }
        fr[i] = f2bf(wr); fz[i] = f2bf(wz); fn[i] = f2bf(wn);
      }
      if (kf < 8) {                 // hidden part: r,z -> regs ; n -> LDS
        Brh[t2][kf] = fr;
        Bzh[t2][kf] = fz;
        AF u; u.v = fn;
        const int idx = ((ct * 8 + kf) * 32 + lane) * 2;
        s_bn[idx] = u.q[0]; s_bn[idx + 1] = u.q[1];
      } else {                      // x part: r,z -> LDS ; n -> regs
        const int kk = kf - 8;
        AF ur; ur.v = fr;
        AF uz; uz.v = fz;
        const int ir = (((ct * 2 + 0) * 4 + kk) * 32 + lane) * 2;
        const int iz = (((ct * 2 + 1) * 4 + kk) * 32 + lane) * 2;
        s_brzx[ir] = ur.q[0]; s_brzx[ir + 1] = ur.q[1];
        s_brzx[iz] = uz.q[0]; s_brzx[iz + 1] = uz.q[1];
        Bnx[t2][kk] = fn;
      }
    }
  }

  // accumulator rows this lane owns: M = v + 8*half
  int lenv[8];
#pragma unroll
  for (int v = 0; v < 8; ++v) lenv[v] = lens[r0 + v + 8*half];

  // x-tile staging: thread -> (row, 8 floats)
  const int xrow  = tid >> 4;          // 0..15
  const int c8    = (tid & 15) * 8;    // 0..120
  const int xbrow = r0 + xrow;
  const int xlen  = lens[xbrow];

  // zero-init hidden state
  {
    unsigned* p32 = (unsigned*)s_h;
    for (int idx = tid; idx < 16 * H_ / 2; idx += 256) p32[idx] = 0u;
  }

  // prefetch x for t = 0
  float4 e0, e1;
  {
    const int tpos0 = dir ? (xlen - 1 < 0 ? 0 : xlen - 1) : 0;
    const int tok = seqs[xbrow * T_ + tpos0];
    e0 = *(const float4*)(embed + tok * E_ + c8);
    e1 = *(const float4*)(embed + tok * E_ + c8 + 4);
  }
  __syncthreads();   // covers weight-fragment stores + h init

  for (int t = 0; t < T_; ++t) {
    const int buf = t & 1;
    // ---- store staged x(t) (one b128 DS store), then prefetch x(t+1) ----
    {
      uint4 pk;
      pk.x = ((unsigned)f2bf_bits(e0.y) << 16) | f2bf_bits(e0.x);
      pk.y = ((unsigned)f2bf_bits(e0.w) << 16) | f2bf_bits(e0.z);
      pk.z = ((unsigned)f2bf_bits(e1.y) << 16) | f2bf_bits(e1.x);
      pk.w = ((unsigned)f2bf_bits(e1.w) << 16) | f2bf_bits(e1.z);
      *(uint4*)&s_x[buf * 2048 + xrow * E_ + c8] = pk;
    }
    {
      const int tn = (t + 1 < T_) ? (t + 1) : t;
      int tpos = dir ? (xlen - 1 - tn) : tn;
      if (tpos < 0) tpos = 0;
      const int tok = seqs[xbrow * T_ + tpos];
      e0 = *(const float4*)(embed + tok * E_ + c8);       // in flight during WMMAs
      e1 = *(const float4*)(embed + tok * E_ + c8 + 4);
    }
    __syncthreads();   // x(t) staged; h(t-1) visible (double buffer => 1 barrier/step)

    // ---- gates = [h|x] @ [W_hh|W_ih]^T for this wave's 2 column tiles ----
    v8f accR[2] = {{}, {}}, accZ[2] = {{}, {}}, accNH[2] = {{}, {}}, accNX[2] = {{}, {}};
#pragma unroll
    for (int kf = 0; kf < 8; ++kf) {       // hidden part, K = 0..255
      AF a;
      const uint4* p = (const uint4*)&s_h[nc * H_ + 32*kf + 8*half];
      a.q[0] = p[0];
      a.q[1] = p[2];
#pragma unroll
      for (int t2 = 0; t2 < 2; ++t2) {
        const int ct = wave + 8 * t2;
        AF bn;
        const int idx = ((ct * 8 + kf) * 32 + lane) * 2;
        bn.q[0] = s_bn[idx]; bn.q[1] = s_bn[idx + 1];
        accR[t2]  = __builtin_amdgcn_wmma_f32_16x16x32_bf16(false, a.v, false, Brh[t2][kf],
                                                            (short)0, accR[t2],  false, false);
        accZ[t2]  = __builtin_amdgcn_wmma_f32_16x16x32_bf16(false, a.v, false, Bzh[t2][kf],
                                                            (short)0, accZ[t2],  false, false);
        accNH[t2] = __builtin_amdgcn_wmma_f32_16x16x32_bf16(false, a.v, false, bn.v,
                                                            (short)0, accNH[t2], false, false);
      }
    }
#pragma unroll
    for (int kk = 0; kk < 4; ++kk) {       // input part, K = 256..383
      AF a;
      const uint4* p = (const uint4*)&s_x[buf * 2048 + nc * E_ + 32*kk + 8*half];
      a.q[0] = p[0];
      a.q[1] = p[2];
#pragma unroll
      for (int t2 = 0; t2 < 2; ++t2) {
        const int ct = wave + 8 * t2;
        AF br, bz;
        const int ir = (((ct * 2 + 0) * 4 + kk) * 32 + lane) * 2;
        const int iz = (((ct * 2 + 1) * 4 + kk) * 32 + lane) * 2;
        br.q[0] = s_brzx[ir]; br.q[1] = s_brzx[ir + 1];
        bz.q[0] = s_brzx[iz]; bz.q[1] = s_brzx[iz + 1];
        accR[t2]  = __builtin_amdgcn_wmma_f32_16x16x32_bf16(false, a.v, false, br.v,
                                                            (short)0, accR[t2],  false, false);
        accZ[t2]  = __builtin_amdgcn_wmma_f32_16x16x32_bf16(false, a.v, false, bz.v,
                                                            (short)0, accZ[t2],  false, false);
        accNX[t2] = __builtin_amdgcn_wmma_f32_16x16x32_bf16(false, a.v, false, Bnx[t2][kk],
                                                            (short)0, accNX[t2], false, false);
      }
    }

    // ---- gate combine + hidden update (no cross-wave exchange needed) ----
#pragma unroll
    for (int t2 = 0; t2 < 2; ++t2) {
      const int j = 16 * (wave + 8 * t2) + nc;
#pragma unroll
      for (int v = 0; v < 8; ++v) {
        const int M = v + 8*half;
        const float r  = sigm(accR[t2][v] + biasR[t2]);
        const float z  = sigm(accZ[t2][v] + biasZ[t2]);
        const float n  = tanhf(accNX[t2][v] + bI_n[t2] + r * (accNH[t2][v] + bH_n[t2]));
        const float ho = bf2f(s_h[M * H_ + j]);
        const float hn = (1.0f - z) * n + z * ho;
        s_h[M * H_ + j] = f2bf_bits(hn);
        if (t == lenv[v] - 1)
          hcat[(r0 + M) * (2*H_) + dir * H_ + j] = hn;   // snapshot last valid step
      }
    }
    // no trailing barrier: next iteration's x-store hits the other buffer; the
    // next __syncthreads() orders s_h writes before the next step's A-loads.
  }
}

// Tiny FC head: [256, 512] @ [512, 2] + b
__global__ void fc_kernel(const float* __restrict__ hcat,
                          const float* __restrict__ Wfc,
                          const float* __restrict__ bfc,
                          float* __restrict__ out)
{
  const int b = blockIdx.x * blockDim.x + threadIdx.x;
  if (b >= B_) return;
  float a0 = bfc[0], a1 = bfc[1];
  const float* h = hcat + b * (2*H_);
  for (int k = 0; k < 2*H_; ++k) {
    const float hv = h[k];
    a0 += hv * Wfc[k];
    a1 += hv * Wfc[2*H_ + k];
  }
  out[2*b]   = a0;
  out[2*b+1] = a1;
}

extern "C" void kernel_launch(void* const* d_in, const int* in_sizes, int n_in,
                              void* d_out, int out_size, void* d_ws, size_t ws_size,
                              hipStream_t stream) {
  const int*   seqs  = (const int*)d_in[0];
  const int*   lens  = (const int*)d_in[1];
  const float* embed = (const float*)d_in[2];
  const float* Wih_f = (const float*)d_in[3];
  const float* Whh_f = (const float*)d_in[4];
  const float* bih_f = (const float*)d_in[5];
  const float* bhh_f = (const float*)d_in[6];
  const float* Wih_b = (const float*)d_in[7];
  const float* Whh_b = (const float*)d_in[8];
  const float* bih_b = (const float*)d_in[9];
  const float* bhh_b = (const float*)d_in[10];
  const float* Wfc   = (const float*)d_in[11];
  const float* bfc   = (const float*)d_in[12];

  float* out  = (float*)d_out;
  float* hcat = (float*)d_ws;                 // B * 2H floats = 512 KB scratch

  // allow 272 KB dynamic LDS (WGP supports up to 320 KB per workgroup)
  (void)hipFuncSetAttribute((const void*)gru_scan_kernel,
                            hipFuncAttributeMaxDynamicSharedMemorySize, LDS_BYTES);

  gru_scan_kernel<<<32, 256, LDS_BYTES, stream>>>(seqs, lens, embed,
      Wih_f, Whh_f, bih_f, bhh_f, Wih_b, Whh_b, bih_b, bhh_b, hcat);
  fc_kernel<<<1, 256, 0, stream>>>(hcat, Wfc, bfc, out);
}
